// LinearAttention_51393578663983
// MI455X (gfx1250) — matrix operand order repaired and checked by reading, not compiled, not run on hardware
//
#include <hip/hip_runtime.h>

// ---------------------------------------------------------------------------
// Linear attention fused pipeline for MI455X (gfx1250, wave32, WMMA bf16).
//   Phase A : KV = X @ Wkv (WMMA), softmax(K) per head, ctx += K^T V (WMMA)
//   Reduce  : sum per-chunk ctx partials (fixed order -> deterministic)
//   Weff    : W_eff[b] = scale * W_q @ (blockdiag(ctx[b]) @ W_out)  (tiny)
//   Phase B : out = X @ W_eff[b] + b_out (WMMA)
// ---------------------------------------------------------------------------

typedef __attribute__((ext_vector_type(16))) __bf16 v16bf;
typedef __attribute__((ext_vector_type(8)))  __bf16 v8bf;
typedef __attribute__((ext_vector_type(8)))  float  v8f;

union V16U { v16bf v; v8bf h[2]; };

#define NTOK   16384   // tokens per batch (128*128)
#define NBATCH 8
#define CCH    64      // phase-A chunks per batch
#define AITER  4       // 64-token tiles per chunk  (64*4*64 = 16384)
#define TP     72      // KTs token-stride (padded, 16B aligned)
#define FP     136     // Vs  feature-stride (padded, 16B aligned)

__device__ __forceinline__ v8f wmma_bf16(v16bf a, v16bf b, v8f c) {
  return __builtin_amdgcn_wmma_f32_16x16x32_bf16(false, a, false, b,
                                                 (short)0, c, false, false);
}

// A-fragment (16x32, M=lane%16): halves 0-7 <- K = kb+8*lhi.., 8-15 <- K = kb+16+8*lhi..
__device__ __forceinline__ v16bf load_a_bf16(const __bf16* p, int lhi) {
  V16U u;
  u.h[0] = *(const v8bf*)(p + 8 * lhi);
  u.h[1] = *(const v8bf*)(p + 16 + 8 * lhi);
  return u.v;
}

// 16 contiguous bf16 elements at p (unused; B fragments are loaded with load_a_bf16)
__device__ __forceinline__ v16bf load_b_bf16(const __bf16* p) {
  V16U u;
  u.h[0] = *(const v8bf*)(p);
  u.h[1] = *(const v8bf*)(p + 8);
  return u.v;
}

// A-fragment sourced from f32 global memory (converted to bf16 on the fly)
__device__ __forceinline__ v16bf load_a_f32(const float* p, int lhi) {
  V16U u;
  const float4 f0 = *(const float4*)(p + 8 * lhi);
  const float4 f1 = *(const float4*)(p + 8 * lhi + 4);
  const float4 f2 = *(const float4*)(p + 16 + 8 * lhi);
  const float4 f3 = *(const float4*)(p + 16 + 8 * lhi + 4);
  u.v[0]  = (__bf16)f0.x; u.v[1]  = (__bf16)f0.y; u.v[2]  = (__bf16)f0.z; u.v[3]  = (__bf16)f0.w;
  u.v[4]  = (__bf16)f1.x; u.v[5]  = (__bf16)f1.y; u.v[6]  = (__bf16)f1.z; u.v[7]  = (__bf16)f1.w;
  u.v[8]  = (__bf16)f2.x; u.v[9]  = (__bf16)f2.y; u.v[10] = (__bf16)f2.z; u.v[11] = (__bf16)f2.w;
  u.v[12] = (__bf16)f3.x; u.v[13] = (__bf16)f3.y; u.v[14] = (__bf16)f3.z; u.v[15] = (__bf16)f3.w;
  return u.v;
}

// --------------------------- prep: Wkv -> bf16 -----------------------------
// WkvB[f'][c] = W_qkv[c][128 + f'],  f' in [0,256): K features then V features
__global__ void kprep(const float* __restrict__ W_qkv, __bf16* __restrict__ WkvB) {
  for (int i = blockIdx.x * blockDim.x + threadIdx.x; i < 128 * 256;
       i += gridDim.x * blockDim.x) {
    int c = i >> 8, f = i & 255;
    WkvB[f * 128 + c] = (__bf16)W_qkv[c * 384 + 128 + f];
  }
}

// --------------------------- phase A ---------------------------------------
__global__ __launch_bounds__(128, 2)
void kphaseA(const float* __restrict__ x, const __bf16* __restrict__ WkvB,
             float* __restrict__ ctxP) {
  __shared__ alignas(16) __bf16 KTs[128 * TP];  // [feature][token]
  __shared__ alignas(16) __bf16 Vs[64 * FP];    // [token][feature]

  const int tid  = threadIdx.x;
  const int w    = tid >> 5;        // wave 0..3
  const int lane = tid & 31;
  const int lrow = lane & 15;
  const int lhi  = lane >> 4;
  const int chunk = blockIdx.x;     // 0..63
  const int b     = blockIdx.y;     // 0..7

  const v8f z8 = {0.f, 0.f, 0.f, 0.f, 0.f, 0.f, 0.f, 0.f};
  v8f cacc[4] = {z8, z8, z8, z8};   // 4 ctx tiles per wave (16 total)

  for (int it = 0; it < AITER; ++it) {
    const int tb = (chunk * AITER + it) * 64;

    // ---- KV = X_tile @ Wkv : each wave owns 16 tokens, 16 N-tiles of 16 ----
    const float* xrow = x + ((size_t)b * NTOK + tb + w * 16 + lrow) * 128;
    v16bf afr[4];
#pragma unroll
    for (int kk = 0; kk < 4; ++kk) afr[kk] = load_a_f32(xrow + kk * 32, lhi);

#pragma unroll 2
    for (int nt = 0; nt < 16; ++nt) {
      v8f acc = z8;
#pragma unroll
      for (int kk = 0; kk < 4; ++kk) {
        v16bf bfr = load_a_bf16(WkvB + (nt * 16 + lrow) * 128 + kk * 32, lhi);
        acc = wmma_bf16(afr[kk], bfr, acc);
      }
      const int f = nt * 16 + lrow;  // output feature (0..255 over K then V)
#pragma unroll
      for (int j = 0; j < 8; ++j) {
        const int m = w * 16 + j + 8 * lhi;  // token within 64-token tile
        if (nt < 8) KTs[f * TP + m] = (__bf16)acc[j];          // K transposed
        else        Vs[m * FP + (f - 128)] = (__bf16)acc[j];   // V natural
      }
    }
    __syncthreads();

    // ---- softmax over head dim (32) per token: thread -> (token, 2 heads) --
    {
      const int token = tid & 63;
      const int h0 = (tid >> 6) * 2;
#pragma unroll
      for (int hh = h0; hh < h0 + 2; ++hh) {
        float vals[32];
        float mx = -1e30f;
#pragma unroll
        for (int d = 0; d < 32; ++d) {
          float v = (float)KTs[(hh * 32 + d) * TP + token];
          vals[d] = v;
          mx = fmaxf(mx, v);
        }
        float s = 0.f;
#pragma unroll
        for (int d = 0; d < 32; ++d) {
          float e = __expf(vals[d] - mx);
          vals[d] = e;
          s += e;
        }
        const float inv = 1.0f / s;
#pragma unroll
        for (int d = 0; d < 32; ++d)
          KTs[(hh * 32 + d) * TP + token] = (__bf16)(vals[d] * inv);
      }
    }
    __syncthreads();

    // ---- ctx_h += K_h^T @ V_h : wave w handles combos 4w..4w+3 -------------
#pragma unroll
    for (int i = 0; i < 4; ++i) {
      const int c  = w * 4 + i;
      const int h  = c >> 2;
      const int dt = (c >> 1) & 1;
      const int et = c & 1;
#pragma unroll
      for (int kk = 0; kk < 2; ++kk) {  // 64 tokens = 2 k-steps of 32
        v16bf ka = load_a_bf16(KTs + (h * 32 + dt * 16 + lrow) * TP + kk * 32, lhi);
        v16bf vb;
#pragma unroll
        for (int s = 0; s < 16; ++s)
          vb[s] = Vs[(kk * 32 + (s & 7) + 8 * lhi + 16 * (s >> 3)) * FP + h * 32 + et * 16 + lrow];
        cacc[i] = wmma_bf16(ka, vb, cacc[i]);
      }
    }
    __syncthreads();
  }

  // ---- write per-chunk ctx partials (plain stores -> deterministic) --------
#pragma unroll
  for (int i = 0; i < 4; ++i) {
    const int c  = w * 4 + i;
    const int h  = c >> 2;
    const int dt = (c >> 1) & 1;
    const int et = c & 1;
#pragma unroll
    for (int j = 0; j < 8; ++j) {
      const int d = dt * 16 + j + 8 * lhi;
      const int e = et * 16 + lrow;
      ctxP[((size_t)b * CCH + chunk) * 4096 + (h * 32 + d) * 32 + e] = cacc[i][j];
    }
  }
}

// --------------------------- ctx reduction ---------------------------------
__global__ void kreduce(const float* __restrict__ ctxP, float* __restrict__ ctxG) {
  const int b = blockIdx.x;
  for (int i = threadIdx.x; i < 4096; i += 256) {
    float s = 0.f;
    for (int c2 = 0; c2 < CCH; ++c2)
      s += ctxP[((size_t)b * CCH + c2) * 4096 + i];
    ctxG[b * 4096 + i] = s;
  }
}

// ---------- W_eff[b] = scale * W_q @ (blockdiag(ctx[b]) @ W_out) ------------
__global__ __launch_bounds__(256)
void kweff(const float* __restrict__ ctxG, const float* __restrict__ W_qkv,
           const float* __restrict__ W_out, __bf16* __restrict__ W_effB) {
  __shared__ float Ms[128 * 128];  // M[f][c]
  const int b = blockIdx.x, tid = threadIdx.x;

  for (int i = tid; i < 128 * 128; i += 256) {
    const int f = i >> 7, c = i & 127;
    const int h = f >> 5, d = f & 31;
    const float* cp = ctxG + (b * 4 + h) * 1024 + d * 32;
    float s = 0.f;
#pragma unroll
    for (int e = 0; e < 32; ++e) s += cp[e] * W_out[(h * 32 + e) * 128 + c];
    Ms[i] = s;
  }
  __syncthreads();

  const float scale = 0.17677669529663687f;  // 32^-0.5
  for (int i = tid; i < 128 * 128; i += 256) {
    const int cin = i >> 7, cout = i & 127;
    float s = 0.f;
#pragma unroll 8
    for (int f = 0; f < 128; ++f) s += W_qkv[cin * 384 + f] * Ms[f * 128 + cout];
    W_effB[(size_t)b * 16384 + cout * 128 + cin] = (__bf16)(s * scale);
  }
}

// --------------------------- phase B: out = X @ W_eff + b_out ---------------
__global__ __launch_bounds__(256, 4)
void kgemm(const float* __restrict__ x, const __bf16* __restrict__ W_effB,
           const float* __restrict__ b_out, float* __restrict__ out) {
  const int tid  = threadIdx.x;
  const int w    = tid >> 5;   // 8 waves
  const int lane = tid & 31;
  const int lrow = lane & 15;
  const int lhi  = lane >> 4;
  const int b  = blockIdx.y;
  const int tb = blockIdx.x * 128;

  const float* xrow = x + ((size_t)b * NTOK + tb + w * 16 + lrow) * 128;
  v16bf afr[4];
#pragma unroll
  for (int kk = 0; kk < 4; ++kk) afr[kk] = load_a_f32(xrow + kk * 32, lhi);

  const __bf16* Wb = W_effB + (size_t)b * 16384;
  const v8f z8 = {0.f, 0.f, 0.f, 0.f, 0.f, 0.f, 0.f, 0.f};

#pragma unroll 4
  for (int nt = 0; nt < 8; ++nt) {
    v8f acc = z8;
#pragma unroll
    for (int kk = 0; kk < 4; ++kk) {
      v16bf bfr = load_a_bf16(Wb + (nt * 16 + lrow) * 128 + kk * 32, lhi);
      acc = wmma_bf16(afr[kk], bfr, acc);
    }
    const float bias = b_out[nt * 16 + lrow];
#pragma unroll
    for (int j = 0; j < 8; ++j) {
      const int token = tb + w * 16 + j + 8 * lhi;
      out[((size_t)b * NTOK + token) * 128 + nt * 16 + lrow] = acc[j] + bias;
    }
  }
}

// ---------------------------------------------------------------------------
extern "C" void kernel_launch(void* const* d_in, const int* in_sizes, int n_in,
                              void* d_out, int out_size, void* d_ws, size_t ws_size,
                              hipStream_t stream) {
  (void)in_sizes; (void)n_in; (void)out_size; (void)ws_size;
  const float* x     = (const float*)d_in[0];
  const float* W_qkv = (const float*)d_in[1];
  const float* W_out = (const float*)d_in[2];
  const float* b_out = (const float*)d_in[3];
  float* out = (float*)d_out;

  char* ws = (char*)d_ws;
  // workspace layout (all 256B-aligned offsets)
  __bf16* WkvB   = (__bf16*)(ws);                                   // 64 KB
  float*  ctxP   = (float*)(ws + 65536);                            // 8 MB
  float*  ctxG   = (float*)(ws + 65536 + (size_t)NBATCH * CCH * 4096 * 4);
  __bf16* W_effB = (__bf16*)(ws + 65536 + (size_t)NBATCH * CCH * 4096 * 4
                                 + (size_t)NBATCH * 4096 * 4);      // 256 KB

  kprep  <<<dim3(64),        128, 0, stream>>>(W_qkv, WkvB);
  kphaseA<<<dim3(CCH, NBATCH), 128, 0, stream>>>(x, WkvB, ctxP);
  kreduce<<<dim3(NBATCH),    256, 0, stream>>>(ctxP, ctxG);
  kweff  <<<dim3(NBATCH),    256, 0, stream>>>(ctxG, W_qkv, W_out, W_effB);
  kgemm  <<<dim3(128, NBATCH), 256, 0, stream>>>(x, W_effB, b_out, out);
}
